// DotProductAttention_9337258901668
// MI455X (gfx1250) — matrix operand order repaired and checked by reading, not compiled, not run on hardware
//
#include <hip/hip_runtime.h>
#include <hip/hip_bf16.h>

// ---------------- problem constants ----------------
static constexpr int BATCH = 32;
static constexpr int SEQ   = 2048;
static constexpr int HD    = 128;   // head dim

static constexpr int WAVES = 8;     // 8 waves = 2 per SIMD32 (latency hiding)
static constexpr int QB    = 16;    // query rows per wave
static constexpr int QW    = WAVES * QB;  // 128 query rows per workgroup
static constexpr int KBLK  = 32;    // keys per block

// LDS padded strides (in shorts); rows 8B-aligned so gathers are b64-pairs
static constexpr int KP = 36;       // sK row stride  (72B rows)
static constexpr int VP = HD + 4;   // sV row stride  (264B rows, 8B aligned)
static constexpr int PP = 36;       // sP row stride  (72B rows)

// ---------------- WMMA fragment types ----------------
typedef __attribute__((ext_vector_type(16))) __bf16  v16bf;
typedef __attribute__((ext_vector_type(8)))  float   v8f;
typedef __attribute__((ext_vector_type(8)))  unsigned int v8u;

union Frag { v8u u; v16bf b; };

// round-to-nearest-even fp32 -> bf16
__device__ __forceinline__ unsigned pk_bf16(float a, float b) {
  union { float f; unsigned u; } ua, ub;
  ua.f = a; ub.f = b;
  unsigned ra = (ua.u + 0x7FFFu + ((ua.u >> 16) & 1u)) >> 16;
  unsigned rb = (ub.u + 0x7FFFu + ((ub.u >> 16) & 1u)) >> 16;
  return (ra & 0xFFFFu) | (rb << 16);
}
__device__ __forceinline__ unsigned short bf16_1(float a) {
  union { float f; unsigned u; } ua; ua.f = a;
  return (unsigned short)((ua.u + 0x7FFFu + ((ua.u >> 16) & 1u)) >> 16);
}

__global__ __launch_bounds__(WAVES * 32, 1)
void fa_fwd_kernel(const float* __restrict__ Qg, const float* __restrict__ Kg,
                   const float* __restrict__ Vg, const int* __restrict__ vlg,
                   const int* __restrict__ causalg, float* __restrict__ Og)
{
  __shared__ unsigned short sK[HD * KP];          // K block, bf16, transposed [d][key]
  __shared__ unsigned short sV[KBLK * VP];        // V block, bf16, row-major [key][d]
  __shared__ unsigned short sP[WAVES][QB * PP];   // per-wave P staging, 16x32 bf16

  const int b    = blockIdx.y;
  const int q0   = blockIdx.x * QW;
  const int tid  = threadIdx.x;
  const int w    = tid >> 5;
  const int lane = tid & 31;
  const int hh   = lane >> 4;    // half-wave (row += 8 in C layout)
  const int l16  = lane & 15;

  const int vl     = vlg[b];
  const int causal = causalg[0];

  const int q0w = q0 + w * QB;
  int kend_wg = causal ? min(q0 + QW, vl) : vl;
  if (kend_wg < 1) kend_wg = 1;
  const int kend_w = causal ? min(q0w + QB, vl) : vl;

  const float scale = 0.08838834764831845f;  // 1/sqrt(128)

  // ---- load Q tile once as 4 A-fragments (16x32 bf16 each) ----
  // lane row = l16; low group d = 32c+8*hh .. +7, high group d = 32c+16+8*hh .. +7
  Frag qf[4];
  const float* qbase = Qg + ((size_t)b * SEQ + (size_t)(q0w + l16)) * HD;
  #pragma unroll
  for (int c = 0; c < 4; ++c) {
    #pragma unroll
    for (int g = 0; g < 2; ++g) {
      int d0 = 32 * c + 16 * g + 8 * hh;
      float4 qa = *(const float4*)(qbase + d0);
      float4 qb = *(const float4*)(qbase + d0 + 4);
      qf[c].u[4 * g + 0] = pk_bf16(qa.x, qa.y);
      qf[c].u[4 * g + 1] = pk_bf16(qa.z, qa.w);
      qf[c].u[4 * g + 2] = pk_bf16(qb.x, qb.y);
      qf[c].u[4 * g + 3] = pk_bf16(qb.z, qb.w);
    }
  }

  v8f acc[8];
  #pragma unroll
  for (int n = 0; n < 8; ++n)
    #pragma unroll
    for (int j = 0; j < 8; ++j) acc[n][j] = 0.0f;

  float m_i[8], l_i[8];
  #pragma unroll
  for (int j = 0; j < 8; ++j) { m_i[j] = -3.0e38f; l_i[j] = 0.0f; }

  const float* kbb = Kg + (size_t)b * SEQ * HD;
  const float* vbb = Vg + (size_t)b * SEQ * HD;

  for (int kb = 0; kb < kend_wg; kb += KBLK) {
    // ---- cooperative load: 32 keys x 128 dims fp32 -> bf16 LDS ----
    #pragma unroll
    for (int i = 0; i < 4; ++i) {
      int idx = tid + 256 * i;      // 0..1023 float4-quads
      int key = idx >> 5;           // 0..31
      int dq  = idx & 31;           // d-quad index
      int kg  = kb + key;
      float4 kx, vx;
      if (kg < SEQ) {
        kx = *(const float4*)(kbb + (size_t)kg * HD + 4 * dq);
        vx = *(const float4*)(vbb + (size_t)kg * HD + 4 * dq);
      } else {
        kx = make_float4(0.f, 0.f, 0.f, 0.f);
        vx = make_float4(0.f, 0.f, 0.f, 0.f);
      }
      sK[(4 * dq + 0) * KP + key] = bf16_1(kx.x);
      sK[(4 * dq + 1) * KP + key] = bf16_1(kx.y);
      sK[(4 * dq + 2) * KP + key] = bf16_1(kx.z);
      sK[(4 * dq + 3) * KP + key] = bf16_1(kx.w);
      *(uint2*)(&sV[key * VP + 4 * dq]) =
          make_uint2(pk_bf16(vx.x, vx.y), pk_bf16(vx.z, vx.w));
    }
    // prefetch next block into caches (global_prefetch_b8)
    if (kb + KBLK < kend_wg) {
      int key = tid >> 5, dq = tid & 31;   // first 8 keys' lines
      __builtin_prefetch(kbb + (size_t)(kb + KBLK + key) * HD + 4 * dq, 0, 0);
      __builtin_prefetch(vbb + (size_t)(kb + KBLK + key) * HD + 4 * dq, 0, 0);
    }
    __syncthreads();

    if (!causal || kb < kend_w) {
      // ---- S = Q K^T : two 16x16 tiles over 4 K-chunks ----
      v8f sc[2];
      #pragma unroll
      for (int t = 0; t < 2; ++t) {
        v8f a;
        #pragma unroll
        for (int j = 0; j < 8; ++j) a[j] = 0.0f;
        #pragma unroll
        for (int c = 0; c < 4; ++c) {
          Frag bk;
          #pragma unroll
          for (int i = 0; i < 8; ++i) {
            int dr = 32 * c + 16 * (i >> 2) + 8 * hh + 2 * (i & 3);
            bk.u[i] = (unsigned)sK[dr * KP + 16 * t + l16] |
                      ((unsigned)sK[(dr + 1) * KP + 16 * t + l16] << 16);
          }
          a = __builtin_amdgcn_wmma_f32_16x16x32_bf16(
                  false, qf[c].b, false, bk.b, (short)0, a, false, false);
        }
        sc[t] = a;
      }

      // ---- mask + online softmax (rows j + 8*hh, cols kb + 16t + l16) ----
      float p0[8], p1[8], em[8];
      #pragma unroll
      for (int j = 0; j < 8; ++j) {
        int rowq = q0w + j + 8 * hh;
        float s0 = sc[0][j] * scale;
        float s1 = sc[1][j] * scale;
        int c0 = kb + l16, c1 = kb + 16 + l16;
        bool msk0 = (c0 >= vl) || (causal && (c0 > rowq));
        bool msk1 = (c1 >= vl) || (causal && (c1 > rowq));
        s0 = msk0 ? -3.0e38f : s0;
        s1 = msk1 ? -3.0e38f : s1;
        float mj = fmaxf(s0, s1);
        #pragma unroll
        for (int off = 8; off > 0; off >>= 1)
          mj = fmaxf(mj, __shfl_xor(mj, off, 32));
        float mnew = fmaxf(m_i[j], mj);
        float e0 = __expf(s0 - mnew);
        float e1 = __expf(s1 - mnew);
        float rs = e0 + e1;
        #pragma unroll
        for (int off = 8; off > 0; off >>= 1)
          rs += __shfl_xor(rs, off, 32);
        float emj = __expf(m_i[j] - mnew);
        l_i[j] = l_i[j] * emj + rs;
        m_i[j] = mnew;
        em[j]  = emj;
        p0[j] = e0; p1[j] = e1;
      }

      // rescale accumulators
      #pragma unroll
      for (int n = 0; n < 8; ++n)
        #pragma unroll
        for (int j = 0; j < 8; ++j) acc[n][j] *= em[j];

      // ---- stage P (C-layout) to LDS, regather as A-fragment ----
      #pragma unroll
      for (int j = 0; j < 8; ++j) {
        sP[w][(j + 8 * hh) * PP + l16]      = bf16_1(p0[j]);
        sP[w][(j + 8 * hh) * PP + 16 + l16] = bf16_1(p1[j]);
      }
      Frag pa;
      #pragma unroll
      for (int g = 0; g < 2; ++g) {
        int k0 = 16 * g + 8 * hh;
        uint2 ta = *(const uint2*)(&sP[w][l16 * PP + k0]);
        uint2 tb = *(const uint2*)(&sP[w][l16 * PP + k0 + 4]);
        pa.u[4 * g + 0] = ta.x;
        pa.u[4 * g + 1] = ta.y;
        pa.u[4 * g + 2] = tb.x;
        pa.u[4 * g + 3] = tb.y;
      }

      // ---- O += P V over 8 d-tiles ----
      #pragma unroll
      for (int n = 0; n < 8; ++n) {
        Frag bv;
        #pragma unroll
        for (int i = 0; i < 8; ++i) {
          int kr = 16 * (i >> 2) + 8 * hh + 2 * (i & 3);
          bv.u[i] = (unsigned)sV[kr * VP + 16 * n + l16] |
                    ((unsigned)sV[(kr + 1) * VP + 16 * n + l16] << 16);
        }
        acc[n] = __builtin_amdgcn_wmma_f32_16x16x32_bf16(
                     false, pa.b, false, bv.b, (short)0, acc[n], false, false);
      }
    }
    __syncthreads();
  }

  // ---- normalize and store ----
  float* obase = Og + ((size_t)b * SEQ + (size_t)q0w) * HD;
  #pragma unroll
  for (int j = 0; j < 8; ++j) {
    float inv = 1.0f / l_i[j];
    #pragma unroll
    for (int n = 0; n < 8; ++n)
      obase[(size_t)(j + 8 * hh) * HD + 16 * n + l16] = acc[n][j] * inv;
  }
}

extern "C" void kernel_launch(void* const* d_in, const int* in_sizes, int n_in,
                              void* d_out, int out_size, void* d_ws, size_t ws_size,
                              hipStream_t stream) {
  const float* Q      = (const float*)d_in[0];
  const float* K      = (const float*)d_in[1];
  const float* V      = (const float*)d_in[2];
  const int*   vlens  = (const int*)d_in[3];
  const int*   causal = (const int*)d_in[4];
  float*       O      = (float*)d_out;

  dim3 grid(SEQ / QW, BATCH, 1);
  dim3 block(WAVES * 32, 1, 1);
  fa_fwd_kernel<<<grid, block, 0, stream>>>(Q, K, V, vlens, causal, O);
}
